// SAGE_31112743092754
// MI455X (gfx1250) — matrix operand order repaired and checked by reading, not compiled, not run on hardware
//
#include <hip/hip_runtime.h>
#include <cstdint>

// ---------------------------------------------------------------------------
// GraphSAGE 3-layer forward for MI455X (gfx1250, wave32).
//   conv:  h = (segsum(x[src])/clip(deg,1)) @ W_l + b_l + x @ W_r
//   trans: GELU(LayerNorm(h))   (layers 0,1)
//   out:   log_softmax(h)       (layer 2)
// GEMMs: v_wmma_f32_16x16x32_bf16 with fragment-order pre-packed bf16 weights
// (global_load_b128) and LDS-staged bf16 activations (ds_load_b128).
// ---------------------------------------------------------------------------

#define N_FEAT 128            // K dimension for every layer
#define LDS_STRIDE 136        // padded bf16 row stride (272B: 16B-aligned, conflict-free)

typedef __attribute__((ext_vector_type(16))) __bf16 v16bf;
typedef __attribute__((ext_vector_type(8)))  __bf16 v8bf;
typedef __attribute__((ext_vector_type(4)))  __bf16 v4bf;
typedef __attribute__((ext_vector_type(8)))  float  v8f;

__device__ __forceinline__ v16bf cat8(v8bf lo, v8bf hi) {
  return __builtin_shufflevector(lo, hi, 0, 1, 2, 3, 4, 5, 6, 7,
                                 8, 9, 10, 11, 12, 13, 14, 15);
}

__device__ __forceinline__ float waveRedSum(float v) {
  #pragma unroll
  for (int off = 16; off > 0; off >>= 1) v += __shfl_xor(v, off, 32);
  return v;
}
__device__ __forceinline__ float waveRedMax(float v) {
  #pragma unroll
  for (int off = 16; off > 0; off >>= 1) v = fmaxf(v, __shfl_xor(v, off, 32));
  return v;
}

// ------------------------------ utility ------------------------------------
__global__ void k_zero(float* __restrict__ p, size_t n) {
  size_t i = (size_t)blockIdx.x * blockDim.x + threadIdx.x;
  if (i < n) p[i] = 0.0f;
}

// ------------------------------ degree -------------------------------------
__global__ void k_deg(const long long* __restrict__ dst, float* __restrict__ deg,
                      int E) {
  int e = blockIdx.x * blockDim.x + threadIdx.x;
  if (e < E) atomicAdd(&deg[(int)dst[e]], 1.0f);
}

// ------------------------- edge scatter-add --------------------------------
// One wave per edge; lane moves 4 consecutive floats (float4 gather,
// 4x global_atomic_add_f32 scatter). Feature matrix is L2-resident (25.6MB).
__global__ void k_scatter(const float* __restrict__ x,
                          const long long* __restrict__ src,
                          const long long* __restrict__ dst,
                          float* __restrict__ msg, int E) {
  int t = blockIdx.x * blockDim.x + threadIdx.x;
  int e = t >> 5;
  int lane = t & 31;
  if (e >= E) return;
  int s = (int)src[e];
  int d = (int)dst[e];
  const float4 v = ((const float4*)(x + (size_t)s * N_FEAT))[lane];
  float* md = msg + (size_t)d * N_FEAT + (lane << 2);
  atomicAdd(md + 0, v.x);
  atomicAdd(md + 1, v.y);
  atomicAdd(md + 2, v.z);
  atomicAdd(md + 3, v.w);
}

// ---------------- pack weights into WMMA B-fragment order ------------------
// B fragment (16-bit 32x16 ISA layout), matching A's table order per lane:
//   k(e) = ks*32 + (lane>>4)*8 + e + (e>=8 ? 8 : 0) ,  col = ct*16 + (lane&15)
// so each lane's fragment is 16 contiguous bf16 (32B). Buffer layout:
//   out[(((ct*4 + ks)*32 + lane)*16 + e]
// Zero-pads cols >= dOut (layer 2: dOut=40 -> 3 col tiles of 16).
__global__ void k_packW(const float* __restrict__ W, __bf16* __restrict__ out,
                        int dOut, int nFrag) {
  int f = blockIdx.x * blockDim.x + threadIdx.x;
  if (f >= nFrag) return;
  int lane = f & 31;
  int ks   = (f >> 5) & 3;
  int ncol = (f >> 7) * 16 + (lane & 15);
  int kbase = ks * 32 + (lane >> 4) * 8;
  bool ok = ncol < dOut;
  __bf16* o = out + ((size_t)f << 4);
  #pragma unroll
  for (int e = 0; e < 16; ++e) {
    float w = ok ? W[(size_t)(kbase + e + (e >= 8 ? 8 : 0)) * dOut + ncol] : 0.0f;
    o[e] = (__bf16)w;
  }
}

// ------------------- fused dual GEMM:  agg@Wl + x@Wr + b -------------------
// Block: 256 threads = 8 waves covering 16 node rows x dOut cols; wave w owns
// the 16x16 tile at cols [16w,16w+16). K loop: 4 x (2 WMMA). Per matrix per
// k-step: A = two ds_load_b128 + shuffle, B = one 32B coalesced global load.
__global__ __launch_bounds__(256)
void k_gemm(const float* __restrict__ msg, const float* __restrict__ deg,
            const float* __restrict__ xin,
            const __bf16* __restrict__ WlP, const float* __restrict__ bias,
            const __bf16* __restrict__ WrP,
            float* __restrict__ hout, int N, int dOut) {
  __shared__ __bf16 sA[16][LDS_STRIDE];  // agg = msg / clip(deg,1)
  __shared__ __bf16 sX[16][LDS_STRIDE];  // x
  const int rowBase = blockIdx.x * 16;
  const int tid = threadIdx.x;

  // ---- stage 16x128 agg and x tiles into LDS as bf16 (float4 granularity) --
  for (int i = tid; i < 16 * 32; i += 256) {
    int r = i >> 5, c4 = (i & 31) << 2;
    int row = rowBase + r;
    v4bf av = {}, xv4 = {};
    if (row < N) {
      float dg = deg[row];
      dg = dg < 1.0f ? 1.0f : dg;
      float rdg = 1.0f / dg;
      float4 mv = *(const float4*)(msg + (size_t)row * N_FEAT + c4);
      float4 xv = *(const float4*)(xin + (size_t)row * N_FEAT + c4);
      av[0] = (__bf16)(mv.x * rdg); av[1] = (__bf16)(mv.y * rdg);
      av[2] = (__bf16)(mv.z * rdg); av[3] = (__bf16)(mv.w * rdg);
      xv4[0] = (__bf16)xv.x; xv4[1] = (__bf16)xv.y;
      xv4[2] = (__bf16)xv.z; xv4[3] = (__bf16)xv.w;
    }
    *(v4bf*)&sA[r][c4] = av;
    *(v4bf*)&sX[r][c4] = xv4;
  }
  __syncthreads();

  const int wave = tid >> 5;
  const int lane = tid & 31;
  const int nColTiles = (dOut + 15) >> 4;
  if (wave >= nColTiles) return;

  const int mrow   = lane & 15;           // A row / B col (lane&15)
  const int khalfA = (lane >> 4) * 8;     // ISA 16-bit A 16x32 layout
  const int ncol   = wave * 16 + mrow;
  const __bf16* aRow = &sA[mrow][0];
  const __bf16* xRow = &sX[mrow][0];

  v8f acc = {};
  #pragma unroll
  for (int ks = 0; ks < 4; ++ks) {
    const int k0 = ks * 32;
    // A fragments: two contiguous 8-bf16 runs per lane (16B-aligned in LDS)
    v16bf afrag = cat8(*(const v8bf*)(aRow + k0 + khalfA),
                       *(const v8bf*)(aRow + k0 + 16 + khalfA));
    v16bf xfrag = cat8(*(const v8bf*)(xRow + k0 + khalfA),
                       *(const v8bf*)(xRow + k0 + 16 + khalfA));
    // B fragments: pre-packed, 32B contiguous per lane, coalesced per wave
    const size_t foff = ((size_t)((wave << 2) | ks) * 32 + lane) << 4;
    v16bf blf = *(const v16bf*)(WlP + foff);
    v16bf brf = *(const v16bf*)(WrP + foff);
    // D = A*B + C   (neg_a, A, neg_b, B, c_mod, C, reuse_a, reuse_b)
    acc = __builtin_amdgcn_wmma_f32_16x16x32_bf16(false, afrag, false, blf,
                                                  (short)0, acc, false, false);
    acc = __builtin_amdgcn_wmma_f32_16x16x32_bf16(false, xfrag, false, brf,
                                                  (short)0, acc, false, false);
  }

  // ---- epilogue: + bias, store (C/D layout: VGPR r -> M=r or r+8) ----
  const bool okCol = ncol < dOut;
  const float bn = okCol ? bias[ncol] : 0.0f;
  const int mtop = (lane >> 4) * 8;
  #pragma unroll
  for (int r = 0; r < 8; ++r) {
    const int row = rowBase + r + mtop;
    if (row < N && okCol) hout[(size_t)row * N_FEAT + ncol] = acc[r] + bn;
  }
}

// --------------------- LayerNorm + exact GELU (wave/node) ------------------
__global__ void k_trans(const float* __restrict__ h, const float* __restrict__ g,
                        const float* __restrict__ be, float* __restrict__ out,
                        int N) {
  int wid  = (blockIdx.x * blockDim.x + threadIdx.x) >> 5;
  int lane = threadIdx.x & 31;
  if (wid >= N) return;
  const float* hr = h + (size_t)wid * N_FEAT;
  float v[4];
  float s = 0.0f;
  #pragma unroll
  for (int i = 0; i < 4; ++i) { v[i] = hr[lane + 32 * i]; s += v[i]; }
  const float mu = waveRedSum(s) * (1.0f / 128.0f);
  float q = 0.0f;
  #pragma unroll
  for (int i = 0; i < 4; ++i) { float d = v[i] - mu; q += d * d; }
  const float rstd = rsqrtf(waveRedSum(q) * (1.0f / 128.0f) + 1e-5f);
  float* outr = out + (size_t)wid * N_FEAT;
  #pragma unroll
  for (int i = 0; i < 4; ++i) {
    int f = lane + 32 * i;
    float xn = (v[i] - mu) * rstd * g[f] + be[f];
    outr[f] = 0.5f * xn * (1.0f + erff(xn * 0.70710678118654752f));
  }
}

// ------------------------- log_softmax (wave/node) -------------------------
__global__ void k_logsoftmax(const float* __restrict__ h, float* __restrict__ out,
                             int N, int dOut) {
  int wid  = (blockIdx.x * blockDim.x + threadIdx.x) >> 5;
  int lane = threadIdx.x & 31;
  if (wid >= N) return;
  const float* hr = h + (size_t)wid * N_FEAT;
  float v0 = (lane < dOut)      ? hr[lane]      : -INFINITY;
  float v1 = (lane + 32 < dOut) ? hr[lane + 32] : -INFINITY;
  float m = waveRedMax(fmaxf(v0, v1));
  float e0 = (lane < dOut)      ? expf(v0 - m) : 0.0f;
  float e1 = (lane + 32 < dOut) ? expf(v1 - m) : 0.0f;
  float lse = m + logf(waveRedSum(e0 + e1));
  if (lane < dOut)      out[(size_t)wid * dOut + lane]      = v0 - lse;
  if (lane + 32 < dOut) out[(size_t)wid * dOut + lane + 32] = v1 - lse;
}

// ---------------------------------------------------------------------------
extern "C" void kernel_launch(void* const* d_in, const int* in_sizes, int n_in,
                              void* d_out, int out_size, void* d_ws, size_t ws_size,
                              hipStream_t stream) {
  const int N = in_sizes[0] / N_FEAT;       // 50000
  const int E = in_sizes[1] / 2;            // 600000

  const float*     x   = (const float*)d_in[0];
  const long long* ei  = (const long long*)d_in[1];   // int64 edge_index [2,E]
  const long long* src = ei;
  const long long* dst = ei + E;

  const float* Wl[3] = { (const float*)d_in[2], (const float*)d_in[5], (const float*)d_in[8] };
  const float* bl[3] = { (const float*)d_in[3], (const float*)d_in[6], (const float*)d_in[9] };
  const float* Wr[3] = { (const float*)d_in[4], (const float*)d_in[7], (const float*)d_in[10] };
  const float* gg[2] = { (const float*)d_in[11], (const float*)d_in[13] };
  const float* bb[2] = { (const float*)d_in[12], (const float*)d_in[14] };
  const int dOut[3]  = { 128, 128, 40 };
  const int nFrag[3] = { 8 * 4 * 32, 8 * 4 * 32, 3 * 4 * 32 };  // colTiles*ksteps*lanes

  // workspace (floats): deg[N] | msg[N*128] | h[N*128] | xbuf[N*128] | packed W (bf16)
  float* deg  = (float*)d_ws;
  float* msg  = deg + N;
  float* hbuf = msg + (size_t)N * N_FEAT;
  float* xbuf = hbuf + (size_t)N * N_FEAT;
  __bf16* wp = (__bf16*)(((uintptr_t)(xbuf + (size_t)N * N_FEAT) + 31) & ~(uintptr_t)31);
  __bf16* WlP[3];
  __bf16* WrP[3];
  {
    __bf16* p = wp;
    for (int l = 0; l < 3; ++l) {
      WlP[l] = p; p += (size_t)nFrag[l] * 16;
      WrP[l] = p; p += (size_t)nFrag[l] * 16;
    }
  }

  const int T = 256;
  const unsigned zb_deg  = (unsigned)((N + T - 1) / T);
  const unsigned zb_msg  = (unsigned)(((size_t)N * N_FEAT + T - 1) / T);
  const unsigned gb_deg  = (unsigned)((E + T - 1) / T);
  const unsigned gb_scat = (unsigned)(((size_t)E * 32 + T - 1) / T);
  const unsigned gb_gemm = (unsigned)((N + 15) / 16);
  const unsigned gb_node = (unsigned)((N + 7) / 8);   // wave per node

  // degree (shared across layers) + weight packing (independent, front-loaded)
  k_zero<<<zb_deg, T, 0, stream>>>(deg, (size_t)N);
  k_deg<<<gb_deg, T, 0, stream>>>(dst, deg, E);
  for (int l = 0; l < 3; ++l) {
    unsigned pb = (unsigned)((nFrag[l] + T - 1) / T);
    k_packW<<<pb, T, 0, stream>>>(Wl[l], WlP[l], dOut[l], nFrag[l]);
    k_packW<<<pb, T, 0, stream>>>(Wr[l], WrP[l], dOut[l], nFrag[l]);
  }

  const float* xcur = x;
  for (int l = 0; l < 3; ++l) {
    k_zero<<<zb_msg, T, 0, stream>>>(msg, (size_t)N * N_FEAT);
    k_scatter<<<gb_scat, T, 0, stream>>>(xcur, src, dst, msg, E);
    k_gemm<<<gb_gemm, T, 0, stream>>>(msg, deg, xcur, WlP[l], bl[l], WrP[l],
                                      hbuf, N, dOut[l]);
    if (l < 2) {
      k_trans<<<gb_node, T, 0, stream>>>(hbuf, gg[l], bb[l], xbuf, N);
      xcur = xbuf;
    }
  }
  k_logsoftmax<<<gb_node, T, 0, stream>>>(hbuf, (float*)d_out, N, dOut[2]);
}
